// ExodusModel_8564164788248
// MI455X (gfx1250) — matrix operand order repaired and checked by reading, not compiled, not run on hardware
//
#include <hip/hip_runtime.h>
#include <hip/hip_bf16.h>

typedef __attribute__((ext_vector_type(16))) _Float16 v16h;
typedef __attribute__((ext_vector_type(8)))  float    v8f;

// ---------------------------------------------------------------------------
// Stage 1: avgpool4 on raw input (n,c,y,x,t) -> pooled0 (n,t,c,64,64)
// t is fastest across threads => every global_load covers 64 consecutive
// floats (256B) -> perfectly coalesced reads of the 268MB input (the
// bandwidth-dominant stage, ~11.5us floor at 23.3 TB/s). Writes are
// 4B-scattered but pooled0 (16.8MB) is absorbed by the 192MB L2.
// ---------------------------------------------------------------------------
__global__ void pool0_kernel(const float* __restrict__ inp,
                             float* __restrict__ out) {
    long tid = (long)blockIdx.x * blockDim.x + threadIdx.x; // 8*2*64*64*64
    int t = (int)(tid & 63);
    long r = tid >> 6;
    int X = (int)(r & 63); r >>= 6;
    int Y = (int)(r & 63); r >>= 6;
    int c = (int)(r & 1);  r >>= 1;
    int n = (int)r;
    const float* base = inp + (((long)(n * 2 + c) * 256 + Y * 4) * 256 + X * 4) * 64 + t;
    __builtin_prefetch(base + (long)4 * 256 * 64, 0, 1); // global_prefetch_b8
    float s = 0.f;
#pragma unroll
    for (int dy = 0; dy < 4; ++dy)
#pragma unroll
        for (int dx = 0; dx < 4; ++dx)
            s += base[((long)dy * 256 + dx) * 64];
    out[((long)((n * 64 + t) * 2 + c) * 64 + Y) * 64 + X] = s * 0.0625f;
}

// ---------------------------------------------------------------------------
// 7x7 pad=3 conv as WMMA GEMM. Branch-free im2col:
//   K padded per channel to 64: k = ic*64 + ky*8 + kx  (ky,kx in [0,8))
//   Padded k entries have ZERO weights in B, so A may read anything finite
//   there -> activations staged in LDS with a zeroed halo (SPAD=HW+8) and
//   every A read is unconditional: two runs of 8 consecutive f32 per K-step.
// Activations reach LDS via global_load_async_to_lds_b32 (ASYNCcnt).
// Each wave owns TWO 16x16 D tiles with independent accumulators so the
// f16 WMMA RAW hazard window is filled with the other tile's work.
// Fragment layouts per ISA 7.12.2:
//   A (16x32 f16): lane=M%16, elem j holds K = 16*(j/8) + 8*(lane/16) + j%8
//   B (32x16 f16): lane=N,    elem j holds K = 16*(j/8) + 8*(lane/16) + j%8
//   D (f32):       VGPR r, lane L -> (M = r + 8*(L/16), N = L%16)
// ---------------------------------------------------------------------------
template<int CIN, int COUT, int HW>
__global__ void conv7_wmma_kernel(const float* __restrict__ in,
                                  const float* __restrict__ w,
                                  float* __restrict__ out) {
    constexpr int KPAD   = CIN * 64;
    constexpr int KSTEPS = KPAD / 32;
    constexpr int NPIX   = HW * HW;
    constexpr int NTILES = NPIX / 16;
    constexpr int TPAIRS = NTILES / 2;
    constexpr int SPAD   = HW + 8;      // rows/cols -3..HW+4 (origin at -3)

    __shared__ float    sIn[CIN * SPAD * SPAD];
    __shared__ _Float16 sW[KPAD * 16];  // sW[k*16 + n], zero padded

    const int img = blockIdx.x;
    const float* imgIn = in + (long)img * CIN * NPIX;

    // Zero the halo cells only (interior is written by the async copy;
    // the two sets are disjoint, so no DS/async ordering hazard).
    for (int i = threadIdx.x; i < CIN * SPAD * SPAD; i += blockDim.x) {
        int cell = i % (SPAD * SPAD);
        int ry = cell / SPAD, rx = cell % SPAD;
        if (!(ry >= 3 && ry < HW + 3 && rx >= 3 && rx < HW + 3))
            sIn[i] = 0.f;
    }
    // Weights, zero-padded into the k = ic*64 + ky*8 + kx space.
    for (int i = threadIdx.x; i < KPAD * 16; i += blockDim.x) {
        int k = i >> 4, n = i & 15;
        int ic = k >> 6, ky = (k >> 3) & 7, kx = k & 7;
        float v = 0.f;
        if (n < COUT && ky < 7 && kx < 7)
            v = w[(n * CIN + ic) * 49 + ky * 7 + kx];
        sW[i] = (_Float16)v;
    }
    // Async-copy the interior image into the haloed LDS tile (ASYNCcnt).
    for (int i = threadIdx.x; i < CIN * NPIX; i += blockDim.x) {
        int ic = i / NPIX;
        int pp = i - ic * NPIX;
        int y = pp / HW, x = pp % HW;
        unsigned ldsOff =
            (unsigned)(unsigned long long)&sIn[(ic * SPAD + y + 3) * SPAD + x + 3];
        const float* g = imgIn + i;
        asm volatile("global_load_async_to_lds_b32 %0, %1, off"
                     :: "v"(ldsOff), "v"(g) : "memory");
    }
    asm volatile("s_wait_asynccnt 0" ::: "memory");
    __syncthreads();

    const int lane  = threadIdx.x & 31;
    const int wv    = threadIdx.x >> 5;
    const int nwv   = blockDim.x >> 5;
    const int half8 = (lane & 16) ? 8 : 0;
    const int col   = lane & 15;          // B/D column (output channel)

    // B fragments are per-wave loop invariants: hoist into registers.
    v16h Bf[KSTEPS];
#pragma unroll
    for (int ks = 0; ks < KSTEPS; ++ks) {
        int kb0 = ks * 32 + ((lane & 16) ? 8 : 0);
#pragma unroll
        for (int j = 0; j < 16; ++j)
            Bf[ks][j] = sW[(kb0 + (j & 7) + ((j >> 3) << 4)) * 16 + col];
    }

    for (int i = wv; i < TPAIRS; i += nwv) {
        const int tt0 = 2 * i, tt1 = 2 * i + 1;
        const int p0 = tt0 * 16 + (lane & 15);
        const int p1 = tt1 * 16 + (lane & 15);
        const int y0 = p0 / HW, x0 = p0 % HW;
        const int y1 = p1 / HW, x1 = p1 % HW;
        v8f acc0 = {}, acc1 = {};
#pragma unroll
        for (int ks = 0; ks < KSTEPS; ++ks) {
            const int kA0 = ks * 32 + half8;   // elems 0..7  (kx = j)
            const int kA1 = kA0 + 16;          // elems 8..15 (kx = j-8)
            const int rowA = (kA0 >> 6) * SPAD + ((kA0 >> 3) & 7);
            const int rowB = (kA1 >> 6) * SPAD + ((kA1 >> 3) & 7);
            const float* r00 = &sIn[(rowA + y0) * SPAD + x0];
            const float* r01 = &sIn[(rowB + y0) * SPAD + x0];
            const float* r10 = &sIn[(rowA + y1) * SPAD + x1];
            const float* r11 = &sIn[(rowB + y1) * SPAD + x1];
            // Phase 1: independent loads into temps (batched DS, one wait).
            float a00[8], a01[8], a10[8], a11[8];
#pragma unroll
            for (int j = 0; j < 8; ++j) {
                a00[j] = r00[j];
                a01[j] = r01[j];
                a10[j] = r10[j];
                a11[j] = r11[j];
            }
            // Phase 2: convert to f16 fragments.
            v16h A0, A1;
#pragma unroll
            for (int j = 0; j < 8; ++j) {
                A0[j]     = (_Float16)a00[j];
                A0[j + 8] = (_Float16)a01[j];
                A1[j]     = (_Float16)a10[j];
                A1[j + 8] = (_Float16)a11[j];
            }
            // Two independent accumulate chains -> hazard NOPs overlap work.
            acc0 = __builtin_amdgcn_wmma_f32_16x16x32_f16(
                false, A0, false, Bf[ks], (short)0, acc0, false, false);
            acc1 = __builtin_amdgcn_wmma_f32_16x16x32_f16(
                false, A1, false, Bf[ks], (short)0, acc1, false, false);
        }
#pragma unroll
        for (int r8 = 0; r8 < 8; ++r8) {
            if (col < COUT) {
                long obase = (long)img * COUT * NPIX + col * NPIX;
                out[obase + tt0 * 16 + r8 + half8] = acc0[r8];
                out[obase + tt1 * 16 + r8 + half8] = acc1[r8];
            }
        }
    }
}

// ---------------------------------------------------------------------------
// IAF with membrane-subtract reset, in place on (n=8, t=64, C, HW, HW).
// One thread per (n,c,y,x) chain; 64 sequential steps.
// ---------------------------------------------------------------------------
template<int C, int HW>
__global__ void iaf_kernel(float* __restrict__ buf) {
    constexpr int PER = C * HW * HW;
    int tid = blockIdx.x * blockDim.x + threadIdx.x;
    if (tid >= 8 * PER) return;
    int n = tid / PER;
    int off = tid - n * PER;
    long base = (long)n * 64 * PER + off;
    float v = 0.f;
    for (int t = 0; t < 64; ++t) {
        v += buf[base + (long)t * PER];
        float s = (v >= 1.0f) ? 1.0f : 0.0f;
        v -= s;
        buf[base + (long)t * PER] = s;
    }
}

// ---------------------------------------------------------------------------
// avgpool4 on (img, C, HW, HW) -> (img, C, HW/4, HW/4)
// ---------------------------------------------------------------------------
template<int C, int HW>
__global__ void pool4_kernel(const float* __restrict__ in,
                             float* __restrict__ out) {
    constexpr int OH = HW / 4;
    long total = (long)512 * C * OH * OH;
    long tid = (long)blockIdx.x * blockDim.x + threadIdx.x;
    if (tid >= total) return;
    int X = (int)(tid % OH); long r = tid / OH;
    int Y = (int)(r % OH);   r /= OH;
    int c = (int)(r % C);    r /= C;
    int img = (int)r;
    const float* base = in + ((long)(img * C + c) * HW + Y * 4) * HW + X * 4;
    float s = 0.f;
#pragma unroll
    for (int dy = 0; dy < 4; ++dy)
#pragma unroll
        for (int dx = 0; dx < 4; ++dx)
            s += base[dy * HW + dx];
    out[tid] = s * 0.0625f;
}

// ---------------------------------------------------------------------------
// feat (512,128) @ wl.T (128,2) -> out (8,2,64).  1024 dot products.
// ---------------------------------------------------------------------------
__global__ void linear_kernel(const float* __restrict__ feat,
                              const float* __restrict__ wl,
                              float* __restrict__ out) {
    int tid = blockIdx.x * blockDim.x + threadIdx.x;
    if (tid >= 1024) return;
    int j = tid & 1;
    int img = tid >> 1;
    float s = 0.f;
#pragma unroll 8
    for (int k = 0; k < 128; ++k)
        s += feat[img * 128 + k] * wl[j * 128 + k];
    int n = img >> 6, t = img & 63;
    out[(n * 2 + j) * 64 + t] = s;
}

extern "C" void kernel_launch(void* const* d_in, const int* in_sizes, int n_in,
                              void* d_out, int out_size, void* d_ws, size_t ws_size,
                              hipStream_t stream) {
    const float* inp = (const float*)d_in[0]; // (8,2,256,256,64)
    const float* w0  = (const float*)d_in[1]; // (4,2,7,7)
    const float* w1  = (const float*)d_in[2]; // (8,4,7,7)
    const float* wl  = (const float*)d_in[3]; // (2,128)
    float* out = (float*)d_out;               // (8,2,64)

    float* ws      = (float*)d_ws;
    float* pooled0 = ws;                // (512,2,64,64)  = 4,194,304 f
    float* h0      = ws + 4194304;      // (512,4,64,64)  = 8,388,608 f
    float* pooled1 = ws + 12582912;     // (512,4,16,16)  =   524,288 f
    float* h1      = ws + 13107200;     // (512,8,16,16)  = 1,048,576 f
    float* feat    = ws + 14155776;     // (512,128)      =    65,536 f
    // total 14,221,312 floats = 56.9 MB of d_ws

    // 1) avgpool4 on raw input (bandwidth-dominant stage)
    pool0_kernel<<<16384, 256, 0, stream>>>(inp, pooled0);
    // 2) conv7 2->4 via WMMA (async-to-LDS staging, 8 waves x 2 tiles)
    conv7_wmma_kernel<2, 4, 64><<<512, 256, 0, stream>>>(pooled0, w0, h0);
    // 3) IAF over t (in place -> spikes)
    iaf_kernel<4, 64><<<512, 256, 0, stream>>>(h0);
    // 4) avgpool4
    pool4_kernel<4, 64><<<2048, 256, 0, stream>>>(h0, pooled1);
    // 5) conv7 4->8 via WMMA
    conv7_wmma_kernel<4, 8, 16><<<512, 256, 0, stream>>>(pooled1, w1, h1);
    // 6) IAF over t (in place -> spikes)
    iaf_kernel<8, 16><<<64, 256, 0, stream>>>(h1);
    // 7) avgpool4 -> flattened feature (layout matches reshape(nt,-1))
    pool4_kernel<8, 16><<<256, 256, 0, stream>>>(h1, feat);
    // 8) final linear -> (8,2,64)
    linear_kernel<<<4, 256, 0, stream>>>(feat, wl, out);
}
